// CausalSelfAttention_85822036509371
// MI455X (gfx1250) — matrix operand order repaired and checked by reading, not compiled, not run on hardware
//
#include <hip/hip_runtime.h>

// Problem constants (from reference)
constexpr int B_  = 4;
constexpr int T_  = 2048;
constexpr int C_  = 1024;
constexpr int NH  = 16;
constexpr int HD  = 64;          // head dim
constexpr int M1  = B_ * T_;     // 8192 rows for the GEMMs

typedef __attribute__((ext_vector_type(16))) _Float16 v16h;
typedef __attribute__((ext_vector_type(8)))  float    v8f;

union Frag { v16h h; unsigned u[8]; _Float16 e[16]; };
union Acc  { v8f  v; float   f[8]; };
union H4   { _Float16 h[4]; uint2 u; };

__device__ __forceinline__ v8f zero8() {
  v8f z = {0.f,0.f,0.f,0.f,0.f,0.f,0.f,0.f};
  return z;
}

__device__ __forceinline__ v8f wmma_f16(v16h a, v16h b, v8f c) {
  // (neg_a, A, neg_b, B, c_mod, C, reuse_a, reuse_b)
  return __builtin_amdgcn_wmma_f32_16x16x32_f16(false, a, false, b, (short)0, c, false, false);
}

// A-matrix fragment (16x32 f16), row-major contiguous k in LDS.
// lane = M%16, hsel = lane/16. k(v,j) = 2v + j + 16*hsel
__device__ __forceinline__ v16h ldfragA(const _Float16* rowbase, int hsel) {
  Frag f;
  const unsigned* p = (const unsigned*)rowbase;   // 4B-aligned (even-half offsets)
#pragma unroll
  for (int v = 0; v < 4; ++v) f.u[v]     = p[v + 8*hsel];
#pragma unroll
  for (int v = 0; v < 4; ++v) f.u[4 + v] = p[4 + v + 8*hsel];
  return f.h;
}

// B-matrix fragment (32x16 f16), LDS holds B transposed [n][k] contiguous in k.
// lane = N%16, k(v,j) = 2v + j + 16*hsel  -> dword index v + 8*hsel
__device__ __forceinline__ v16h ldfragB(const _Float16* rowbase, int hsel) {
  Frag f;
  const unsigned* p = (const unsigned*)rowbase;
#pragma unroll
  for (int v = 0; v < 8; ++v) f.u[v] = p[v + 8*hsel];
  return f.h;
}

// -------- CDNA5 async LDS copy (ASYNCcnt) --------------------------------
// LDS dest address comes from the vdst VGPR plus offset; 16B chunks, LDS dst must be 16B aligned.
__device__ __forceinline__ void async_b128(unsigned lds_off, const _Float16* g) {
  asm volatile("global_load_async_to_lds_b128 %0, %1, off"
               :: "v"(lds_off), "v"((unsigned long long)(const void*)g)
               : "memory");
}
__device__ __forceinline__ void wait_async0() {
  asm volatile("s_wait_asynccnt 0x0" ::: "memory");
}
__device__ __forceinline__ unsigned lds_addr(const void* p) {
  return (unsigned)(unsigned long long)p;   // low 32 bits of flat LDS addr = LDS offset
}

// ---------------------------------------------------------------------------
// Prologue kernels: fp32 -> f16 convert, and transpose-convert for weights.
// ---------------------------------------------------------------------------
__launch_bounds__(256)
__global__ void cvt_f16(const float* __restrict__ in, _Float16* __restrict__ out, int n4) {
  int i = blockIdx.x * 256 + threadIdx.x;
  if (i < n4) {
    float4 v = ((const float4*)in)[i];
    H4 h;
    h.h[0] = (_Float16)v.x; h.h[1] = (_Float16)v.y;
    h.h[2] = (_Float16)v.z; h.h[3] = (_Float16)v.w;
    ((uint2*)out)[i] = h.u;
  }
}

// out[c][r] = (f16) in[r][c];  in: [R][Cc] fp32.  grid (Cc/64, R/64), 256 thr.
__launch_bounds__(256)
__global__ void transpose_cvt(const float* __restrict__ in, _Float16* __restrict__ out,
                              int R, int Cc) {
  __shared__ _Float16 t[64 * 65];
  const int r0 = blockIdx.y * 64, c0 = blockIdx.x * 64;
#pragma unroll
  for (int it = 0; it < 16; ++it) {
    int e = threadIdx.x + it * 256;
    int r = e >> 6, c = e & 63;
    t[c * 65 + r] = (_Float16)in[(size_t)(r0 + r) * Cc + c0 + c];
  }
  __syncthreads();
#pragma unroll
  for (int it = 0; it < 16; ++it) {
    int e = threadIdx.x + it * 256;
    int c = e >> 6, r = e & 63;
    out[(size_t)(c0 + c) * R + r0 + r] = t[c * 65 + r];
  }
}

// ---------------------------------------------------------------------------
// f16 GEMM core: 256 thr (8 waves), 128x128 tile, k-step 32, double-buffered
// async-to-LDS staging.  A: [M][K] f16 row-major.  B: [N][K] f16 (transposed).
// LDS row stride 40 halfs = 80B (16B aligned rows; 20-dword bank stride).
// ---------------------------------------------------------------------------
constexpr int LDA = 40;

// ---------------- Kernel 1: qkv = xh @ WaT^T + b, scatter to q/k/v [B,H,T,D]
__launch_bounds__(256)
__global__ void qkv_gemm(const _Float16* __restrict__ xh, const _Float16* __restrict__ WaT,
                         const float* __restrict__ ba,
                         _Float16* __restrict__ qh, _Float16* __restrict__ kh,
                         _Float16* __restrict__ vh) {
  __shared__ _Float16 As[2][128 * LDA];
  __shared__ _Float16 Bs[2][128 * LDA];

  const int tid  = threadIdx.x;
  const int lane = tid & 31, wave = tid >> 5;
  const int lrow = lane & 15, hsel = lane >> 4;
  const int m0 = blockIdx.y * 128;
  const int n0 = blockIdx.x * 128;
  const int mw = (wave >> 1) * 32;
  const int nw = (wave &  1) * 64;

  const unsigned aB[2] = { lds_addr(As[0]), lds_addr(As[1]) };
  const unsigned bB[2] = { lds_addr(Bs[0]), lds_addr(Bs[1]) };

  auto stage = [&](int buf, int k0) {
#pragma unroll
    for (int it = 0; it < 2; ++it) {
      int e = tid + it * 256;          // chunk id 0..511 (4 chunks/row of 32 halfs)
      int r = e >> 2, c = (e & 3) * 8;
      unsigned lo = (unsigned)(r * LDA + c) * 2u;
      async_b128(aB[buf] + lo, xh  + (size_t)(m0 + r) * C_ + k0 + c);
      async_b128(bB[buf] + lo, WaT + (size_t)(n0 + r) * C_ + k0 + c);
    }
  };

  Acc acc[2][4];
#pragma unroll
  for (int i = 0; i < 2; ++i)
#pragma unroll
    for (int j = 0; j < 4; ++j) acc[i][j].v = zero8();

  stage(0, 0);
  int buf = 0;
  for (int k0 = 0; k0 < C_; k0 += 32) {
    wait_async0();
    __syncthreads();
    if (k0 + 32 < C_) stage(buf ^ 1, k0 + 32);

    v16h afr[2], bfr[4];
#pragma unroll
    for (int i = 0; i < 2; ++i) afr[i] = ldfragA(&As[buf][(mw + i*16 + lrow) * LDA], hsel);
#pragma unroll
    for (int j = 0; j < 4; ++j) bfr[j] = ldfragB(&Bs[buf][(nw + j*16 + lrow) * LDA], hsel);
#pragma unroll
    for (int i = 0; i < 2; ++i)
#pragma unroll
      for (int j = 0; j < 4; ++j)
        acc[i][j].v = wmma_f16(afr[i], bfr[j], acc[i][j].v);
    buf ^= 1;
  }

  // epilogue: + bias, scatter into q/k/v [B,H,T,D] f16
#pragma unroll
  for (int j = 0; j < 4; ++j) {
    int nb = n0 + nw + j * 16 + lrow;          // 0..3071
    float bias = ba[nb];
    int part = nb >> 10, cc = nb & 1023;
    int h = cc >> 6, d = cc & 63;
    _Float16* dst = (part == 0) ? qh : (part == 1) ? kh : vh;
#pragma unroll
    for (int i = 0; i < 2; ++i) {
#pragma unroll
      for (int r = 0; r < 8; ++r) {
        int gm = m0 + mw + i * 16 + r + 8 * hsel;
        int b = gm >> 11, t = gm & (T_ - 1);
        dst[(((size_t)(b * NH + h)) * T_ + t) * HD + d] = (_Float16)(acc[i][j].f[r] + bias);
      }
    }
  }
}

// ---------------------------------------------------------------------------
// Kernel 2: flash attention.  128 thr (4 waves) per (b, h, 64-row q tile).
// Async-to-LDS staged Q/K/V tiles; online softmax; S and P@V via WMMA.
// ---------------------------------------------------------------------------
constexpr int LQ = 72;   // halfs; 144B row stride (16B aligned rows)

__launch_bounds__(128)
__global__ void flash_attn(const _Float16* __restrict__ qh, const _Float16* __restrict__ kh,
                           const _Float16* __restrict__ vh, _Float16* __restrict__ oh) {
  __shared__ _Float16 Qs[64 * LQ];
  __shared__ _Float16 Ks[64 * LQ];
  __shared__ _Float16 Vs[64 * LQ];
  __shared__ _Float16 Ps[4 * 16 * LQ];

  const int tid  = threadIdx.x;
  const int lane = tid & 31, wave = tid >> 5;
  const int lrow = lane & 15, hsel = lane >> 4;
  const int qt = blockIdx.x, h = blockIdx.y, b = blockIdx.z;
  const size_t base = ((size_t)(b * NH + h)) * T_ * HD;
  const int q0 = qt * 64;
  const int qrow_base = q0 + wave * 16;
  const float scale = 0.125f;           // 1/sqrt(64)

  const unsigned qsB = lds_addr(Qs), ksB = lds_addr(Ks), vsB = lds_addr(Vs);

  // stage Q tile once (async)
#pragma unroll
  for (int it = 0; it < 4; ++it) {
    int e = tid + it * 128;            // chunk 0..511 (8 chunks/row of 64 halfs)
    int r = e >> 3, c = (e & 7) * 8;
    async_b128(qsB + (unsigned)(r * LQ + c) * 2u, qh + base + (size_t)(q0 + r) * HD + c);
  }
  wait_async0();
  __syncthreads();
  v16h qf[2];
#pragma unroll
  for (int ks = 0; ks < 2; ++ks)
    qf[ks] = ldfragA(Qs + (wave * 16 + lrow) * LQ + ks * 32, hsel);

  Acc oacc[4];
#pragma unroll
  for (int j = 0; j < 4; ++j) oacc[j].v = zero8();
  float mrow[8], lsum[8];
#pragma unroll
  for (int r = 0; r < 8; ++r) { mrow[r] = -3.0e38f; lsum[r] = 0.f; }

  for (int j0 = 0; j0 <= q0; j0 += 64) {
    __syncthreads();                   // everyone done reading previous K/V tile
#pragma unroll
    for (int it = 0; it < 4; ++it) {
      int e = tid + it * 128;
      int r = e >> 3, c = (e & 7) * 8;
      unsigned lo = (unsigned)(r * LQ + c) * 2u;
      async_b128(ksB + lo, kh + base + (size_t)(j0 + r) * HD + c);
      async_b128(vsB + lo, vh + base + (size_t)(j0 + r) * HD + c);
    }
    if (j0 + 64 <= q0) {               // prefetch next tiles (global_prefetch_b8)
      __builtin_prefetch(kh + base + (size_t)(j0 + 64) * HD + tid * 32, 0, 1);
      __builtin_prefetch(vh + base + (size_t)(j0 + 64) * HD + tid * 32, 0, 1);
    }
    wait_async0();
    __syncthreads();

    // S = Q K^T  (B fragment: lane = key col j, k-dim = d; Ks rows contiguous in d)
    Acc s[4];
#pragma unroll
    for (int jt = 0; jt < 4; ++jt) s[jt].v = zero8();
#pragma unroll
    for (int ks = 0; ks < 2; ++ks) {
#pragma unroll
      for (int jt = 0; jt < 4; ++jt) {
        v16h bk = ldfragB(Ks + (jt * 16 + lrow) * LQ + ks * 32, hsel);
        s[jt].v = wmma_f16(qf[ks], bk, s[jt].v);
      }
    }

    // scale + causal mask + online softmax (row stats per half-wave)
    float rmax[8], fs[8], rsum[8];
#pragma unroll
    for (int r = 0; r < 8; ++r) {
      int qrow = qrow_base + r + 8 * hsel;
      float mx = -3.0e38f;
#pragma unroll
      for (int jt = 0; jt < 4; ++jt) {
        int kcol = j0 + jt * 16 + lrow;
        float val = s[jt].f[r] * scale;
        val = (kcol <= qrow) ? val : -3.0e38f;
        s[jt].f[r] = val;
        mx = fmaxf(mx, val);
      }
      rmax[r] = mx;
    }
#pragma unroll
    for (int r = 0; r < 8; ++r) {
#pragma unroll
      for (int m = 1; m < 16; m <<= 1)
        rmax[r] = fmaxf(rmax[r], __shfl_xor(rmax[r], m, 32));
      float mnew = fmaxf(mrow[r], rmax[r]);
      fs[r] = __expf(mrow[r] - mnew);
      mrow[r] = mnew;
    }
#pragma unroll
    for (int r = 0; r < 8; ++r) {
      float sum = 0.f;
#pragma unroll
      for (int jt = 0; jt < 4; ++jt) {
        float p = __expf(s[jt].f[r] - mrow[r]);
        s[jt].f[r] = p;
        sum += p;
      }
      rsum[r] = sum;
    }
#pragma unroll
    for (int r = 0; r < 8; ++r) {
#pragma unroll
      for (int m = 1; m < 16; m <<= 1) rsum[r] += __shfl_xor(rsum[r], m, 32);
      lsum[r] = lsum[r] * fs[r] + rsum[r];
    }
#pragma unroll
    for (int dt = 0; dt < 4; ++dt)
#pragma unroll
      for (int r = 0; r < 8; ++r) oacc[dt].f[r] *= fs[r];

    // transpose P (D-layout) -> per-wave LDS rows for A-fragment reload
    _Float16* Pw = Ps + wave * 16 * LQ;
#pragma unroll
    for (int jt = 0; jt < 4; ++jt)
#pragma unroll
      for (int r = 0; r < 8; ++r)
        Pw[(r + 8 * hsel) * LQ + jt * 16 + lrow] = (_Float16)s[jt].f[r];

    // O += P @ V   (B fragment strided: k = key row in Vs, col = d)
#pragma unroll
    for (int ks = 0; ks < 2; ++ks) {
      v16h ap = ldfragA(Pw + lrow * LQ + ks * 32, hsel);
#pragma unroll
      for (int dt = 0; dt < 4; ++dt) {
        Frag f;
#pragma unroll
        for (int v = 0; v < 8; ++v) {
          int k = 2 * v + 16 * hsel + ks * 32;
          f.e[2 * v]     = Vs[(k    ) * LQ + dt * 16 + lrow];
          f.e[2 * v + 1] = Vs[(k + 1) * LQ + dt * 16 + lrow];
        }
        oacc[dt].v = wmma_f16(ap, f.h, oacc[dt].v);
      }
    }
  }

  // normalize, store O f16 in [B,T,C] layout (proj GEMM A operand)
#pragma unroll
  for (int dt = 0; dt < 4; ++dt) {
    int d = dt * 16 + lrow;
#pragma unroll
    for (int r = 0; r < 8; ++r) {
      int qrow = qrow_base + r + 8 * hsel;
      float o = oacc[dt].f[r] / lsum[r];
      oh[((size_t)(b * T_ + qrow)) * C_ + h * HD + d] = (_Float16)o;
    }
  }
}

// ---------------- Kernel 3: out = O @ WpT^T + b_proj (fp32 out) ------------
__launch_bounds__(256)
__global__ void proj_gemm(const _Float16* __restrict__ oh, const _Float16* __restrict__ WpT,
                          const float* __restrict__ bp, float* __restrict__ out) {
  __shared__ _Float16 As[2][128 * LDA];
  __shared__ _Float16 Bs[2][128 * LDA];

  const int tid  = threadIdx.x;
  const int lane = tid & 31, wave = tid >> 5;
  const int lrow = lane & 15, hsel = lane >> 4;
  const int m0 = blockIdx.y * 128;
  const int n0 = blockIdx.x * 128;
  const int mw = (wave >> 1) * 32;
  const int nw = (wave &  1) * 64;

  const unsigned aB[2] = { lds_addr(As[0]), lds_addr(As[1]) };
  const unsigned bB[2] = { lds_addr(Bs[0]), lds_addr(Bs[1]) };

  auto stage = [&](int buf, int k0) {
#pragma unroll
    for (int it = 0; it < 2; ++it) {
      int e = tid + it * 256;
      int r = e >> 2, c = (e & 3) * 8;
      unsigned lo = (unsigned)(r * LDA + c) * 2u;
      async_b128(aB[buf] + lo, oh  + (size_t)(m0 + r) * C_ + k0 + c);
      async_b128(bB[buf] + lo, WpT + (size_t)(n0 + r) * C_ + k0 + c);
    }
  };

  Acc acc[2][4];
#pragma unroll
  for (int i = 0; i < 2; ++i)
#pragma unroll
    for (int j = 0; j < 4; ++j) acc[i][j].v = zero8();

  stage(0, 0);
  int buf = 0;
  for (int k0 = 0; k0 < C_; k0 += 32) {
    wait_async0();
    __syncthreads();
    if (k0 + 32 < C_) stage(buf ^ 1, k0 + 32);

    v16h afr[2], bfr[4];
#pragma unroll
    for (int i = 0; i < 2; ++i) afr[i] = ldfragA(&As[buf][(mw + i*16 + lrow) * LDA], hsel);
#pragma unroll
    for (int j = 0; j < 4; ++j) bfr[j] = ldfragB(&Bs[buf][(nw + j*16 + lrow) * LDA], hsel);
#pragma unroll
    for (int i = 0; i < 2; ++i)
#pragma unroll
      for (int j = 0; j < 4; ++j)
        acc[i][j].v = wmma_f16(afr[i], bfr[j], acc[i][j].v);
    buf ^= 1;
  }

#pragma unroll
  for (int j = 0; j < 4; ++j) {
    int nb = n0 + nw + j * 16 + lrow;
    float bias = bp[nb];
#pragma unroll
    for (int i = 0; i < 2; ++i) {
#pragma unroll
      for (int r = 0; r < 8; ++r) {
        int gm = m0 + mw + i * 16 + r + 8 * hsel;
        out[(size_t)gm * C_ + nb] = acc[i][j].f[r] + bias;
      }
    }
  }
}

// ---------------------------------------------------------------------------
extern "C" void kernel_launch(void* const* d_in, const int* in_sizes, int n_in,
                              void* d_out, int out_size, void* d_ws, size_t ws_size,
                              hipStream_t stream) {
  const float* x  = (const float*)d_in[0];
  const float* Wa = (const float*)d_in[1];
  const float* ba = (const float*)d_in[2];
  const float* Wp = (const float*)d_in[3];
  const float* bp = (const float*)d_in[4];
  float* out = (float*)d_out;

  const size_t S = (size_t)M1 * C_;            // 8 Mi halfs
  _Float16* ws  = (_Float16*)d_ws;
  _Float16* xh  = ws;                          // [M1][C] f16 ; dead after qkv_gemm
  _Float16* oh  = ws;                          // aliases xh (flash output / proj A)
  _Float16* qh  = ws + S;
  _Float16* kh  = ws + 2 * S;
  _Float16* vh  = ws + 3 * S;
  _Float16* WaT = ws + 4 * S;                  // [3072][1024] f16
  _Float16* WpT = WaT + (size_t)3072 * C_;     // [1024][1024] f16
  // total: 36 Mi halfs = 72 MB of d_ws

  cvt_f16      <<<(int)(S / 4 / 256), 256, 0, stream>>>(x, xh, (int)(S / 4));
  transpose_cvt<<<dim3(3072 / 64, C_ / 64), 256, 0, stream>>>(Wa, WaT, C_, 3072);
  transpose_cvt<<<dim3(C_   / 64, C_ / 64), 256, 0, stream>>>(Wp, WpT, C_, C_);

  qkv_gemm  <<<dim3(3072 / 128, M1 / 128), 256, 0, stream>>>(xh, WaT, ba, qh, kh, vh);
  flash_attn<<<dim3(T_ / 64, NH, B_),      128, 0, stream>>>(qh, kh, vh, oh);
  proj_gemm <<<dim3(C_ / 128, M1 / 128),   256, 0, stream>>>(oh, WpT, bp, out);
}
